// _SelfAttAxialRowmajor_50302656971165
// MI455X (gfx1250) — matrix operand order repaired and checked by reading, not compiled, not run on hardware
//
#include <hip/hip_runtime.h>
#include <hip/hip_bf16.h>

// Problem constants (from reference): B=4, S=4096, D=1024, H=16, HD=64, SEG=NS=64
#define BQ   4
#define SQ   4096
#define DQ   1024
#define HQ   16
#define HDQ  64
#define SEGQ 64
#define NSQ  64
#define BSQ  (BQ * SQ)                 // 16384 rows
#define NELQ ((size_t)BSQ * DQ)        // 16,777,216 elements

typedef __attribute__((ext_vector_type(16))) _Float16 v16h;
typedef __attribute__((ext_vector_type(8)))  float    v8f;
typedef __attribute__((ext_vector_type(4)))  unsigned int v4u;
typedef __attribute__((ext_vector_type(4)))  int      v4i;
typedef __attribute__((ext_vector_type(8)))  int      v8i;

#if __has_builtin(__builtin_amdgcn_tensor_load_to_lds) && \
    __has_builtin(__builtin_amdgcn_s_wait_tensorcnt)
#define HAVE_TDM 1
#else
#define HAVE_TDM 0
#endif

#if HAVE_TDM
// Issue one TDM load of a 2D f16 tile (rows x cols, row stride in elements)
// from global memory into LDS (packed row-major, cols contiguous).
// D# layout per CDNA5 ISA ch.8: group0 = {flags, lds_addr, global_addr, type=2},
// group1 = {data_size, tensor dims, tile dims, dim0 stride}. 2D -> groups 2/3 zero.
__device__ __forceinline__ void tdm_load_2d_f16(const _Float16* gsrc,
                                                void* lds_dst, unsigned cols,
                                                unsigned rows,
                                                unsigned long long stride_elems) {
  const unsigned long long ga = (unsigned long long)gsrc;
  const unsigned lds = (unsigned)(unsigned long long)lds_dst;
  v4u g0;
  g0[0] = 1u;                                   // count=1 valid descriptor
  g0[1] = lds;                                  // lds_addr (bytes)
  g0[2] = (unsigned)ga;                         // global_addr[31:0]
  g0[3] = ((unsigned)(ga >> 32) & 0x01FFFFFFu)  // global_addr[56:32]
          | (2u << 30);                         // type = 2 ("image")
  v8i g1;
  g1[0] = (int)(1u << 16);                      // data_size=1 (2 bytes), no multicast
  g1[1] = (int)((cols & 0xFFFFu) << 16);        // tensor_dim0[15:0] @ bits 63:48
  g1[2] = (int)(((cols >> 16) & 0xFFFFu) |      // tensor_dim0[31:16]
                ((rows & 0xFFFFu) << 16));      // tensor_dim1[15:0]
  g1[3] = (int)(((rows >> 16) & 0xFFFFu) |      // tensor_dim1[31:16]
                ((cols & 0xFFFFu) << 16));      // tile_dim0 = cols
  g1[4] = (int)(rows & 0xFFFFu);                // tile_dim1 = rows, tile_dim2 = 0
  g1[5] = (int)(unsigned)stride_elems;          // tensor_dim0_stride[31:0]
  g1[6] = (int)((unsigned)(stride_elems >> 32) & 0xFFFFu);  // stride[47:32]
  g1[7] = 0;                                    // tensor_dim1_stride hi (unused, 2D)
  const v4i z = {0, 0, 0, 0};
#if defined(__clang_major__) && (__clang_major__ >= 23)
  const v8i z8 = {0, 0, 0, 0, 0, 0, 0, 0};
  __builtin_amdgcn_tensor_load_to_lds(g0, g1, z, z, z8, 0);
#else
  __builtin_amdgcn_tensor_load_to_lds(g0, g1, z, z, 0);
#endif
}
#define TDM_WAIT0() __builtin_amdgcn_s_wait_tensorcnt((short)0)
#endif  // HAVE_TDM

// ---------------- WMMA fragment helpers (wave32, contiguous k = 16*half + s in A and B) ----------------
// A 16x32 f16 (MxK): lanes 0-15 M=lane, K=0..15;
// lanes 16-31: K=16..31.
__device__ __forceinline__ v16h load_a_frag(const _Float16* t, int ld) {
  const int lane = threadIdx.x & 31;
  const int m = lane & 15, half = lane >> 4;
  v16h a;
#pragma unroll
  for (int j = 0; j < 8; ++j) {
    const int kb = 16 * half + 2 * j;
    a[2 * j]     = t[m * ld + kb];
    a[2 * j + 1] = t[m * ld + kb + 1];
  }
  return a;
}
// B 32x16 f16 (KxN), row-major KxN tile: lanes 0-15 K=0..15, lanes 16-31 K=16..31.
__device__ __forceinline__ v16h load_b_frag(const _Float16* t, int ld) {
  const int lane = threadIdx.x & 31;
  const int n = lane & 15, half = lane >> 4;
  v16h b;
#pragma unroll
  for (int j = 0; j < 8; ++j) {
    const int k0 = 16 * half + 2 * j;
    b[2 * j]     = t[k0 * ld + n];
    b[2 * j + 1] = t[(k0 + 1) * ld + n];
  }
  return b;
}
// B fragment from an N x K (row-major) tile, i.e. B[k][n] = tile[n][k] (Q*K^T)
__device__ __forceinline__ v16h load_bT_frag(const _Float16* t, int ld) {
  const int lane = threadIdx.x & 31;
  const int n = lane & 15, half = lane >> 4;
  v16h b;
#pragma unroll
  for (int j = 0; j < 8; ++j) {
    const int k0 = 16 * half + 2 * j;
    b[2 * j]     = t[n * ld + k0];
    b[2 * j + 1] = t[n * ld + k0 + 1];
  }
  return b;
}
__device__ __forceinline__ v8f wmma16(v16h a, v16h b, v8f c) {
  return __builtin_amdgcn_wmma_f32_16x16x32_f16(false, a, false, b, (short)0, c,
                                                false, false);
}

// ------------------------- f32 -> f16 conversion -------------------------
__global__ __launch_bounds__(256) void cvt_f16_kernel(
    const float* __restrict__ s, _Float16* __restrict__ d, int n) {
  const int i = blockIdx.x * 256 + threadIdx.x;
  if (i < n) d[i] = (_Float16)s[i];
}

// ---------------- Generic tiled GEMM: C(MxN) = A(MxK) * B(KxN), f16 in ----------------
// Block tile 128x64, 256 threads = 8 waves, each wave does a 16x64 row slice.
// TDM double-buffered staging: wave 0 DMAs the next A/B tiles into LDS while
// all waves run WMMA on the current tiles.
template <bool OUT_F16, bool RESID>
__global__ __launch_bounds__(256) void gemm_kernel(
    const _Float16* __restrict__ Ah, const _Float16* __restrict__ Bh,
    void* __restrict__ outp, const float* __restrict__ resid, float scale,
    int Kd, int N) {
  __shared__ _Float16 As[2][128 * 32];
  __shared__ _Float16 Bs[2][32 * 64];
  const int tid = threadIdx.x, lane = tid & 31, w = tid >> 5;
  const int row0 = blockIdx.y * 128;
  const int col0 = blockIdx.x * 64;
  v8f acc[4] = {{}, {}, {}, {}};

#if HAVE_TDM
  if (tid < 32) {  // wave 0 drives the tensor DMA
    tdm_load_2d_f16(Ah + (size_t)row0 * Kd, &As[0][0], 32, 128,
                    (unsigned long long)Kd);
    tdm_load_2d_f16(Bh + (size_t)col0, &Bs[0][0], 64, 32,
                    (unsigned long long)N);
    TDM_WAIT0();
  }
  __syncthreads();
  int cur = 0;
  for (int kk = 0; kk < Kd; kk += 32) {
    const int nxt = cur ^ 1;
    if (kk + 32 < Kd && tid < 32) {  // prefetch next tiles via TDM
      tdm_load_2d_f16(Ah + (size_t)row0 * Kd + (kk + 32), &As[nxt][0], 32, 128,
                      (unsigned long long)Kd);
      tdm_load_2d_f16(Bh + (size_t)(kk + 32) * N + col0, &Bs[nxt][0], 64, 32,
                      (unsigned long long)N);
    }
    const v16h a = load_a_frag(&As[cur][w * 16 * 32], 32);
#pragma unroll
    for (int nt = 0; nt < 4; ++nt) {
      const v16h b = load_b_frag(&Bs[cur][nt * 16], 64);
      acc[nt] = wmma16(a, b, acc[nt]);
    }
    __syncthreads();                 // everyone done reading cur
    if (tid < 32) TDM_WAIT0();       // next tiles landed in LDS
    __syncthreads();
    cur = nxt;
  }
#else
  for (int kk = 0; kk < Kd; kk += 32) {
    __syncthreads();
#pragma unroll
    for (int i = 0; i < 16; ++i) {  // A tile: 128x32
      const int e = tid + i * 256;
      const int r = e >> 5, c = e & 31;
      As[0][e] = Ah[(size_t)(row0 + r) * Kd + kk + c];
    }
#pragma unroll
    for (int i = 0; i < 8; ++i) {  // B tile: 32x64
      const int e = tid + i * 256;
      const int r = e >> 6, c = e & 63;
      Bs[0][e] = Bh[(size_t)(kk + r) * N + col0 + c];
    }
    __syncthreads();
    const v16h a = load_a_frag(&As[0][w * 16 * 32], 32);
#pragma unroll
    for (int nt = 0; nt < 4; ++nt) {
      const v16h b = load_b_frag(&Bs[0][nt * 16], 64);
      acc[nt] = wmma16(a, b, acc[nt]);
    }
  }
#endif

  const int n = lane & 15, mb = (lane >> 4) * 8;
#pragma unroll
  for (int nt = 0; nt < 4; ++nt) {
#pragma unroll
    for (int i = 0; i < 8; ++i) {
      const int r = row0 + w * 16 + mb + i;
      const int c = col0 + nt * 16 + n;
      float v = acc[nt][i] * scale;
      if constexpr (RESID) v += resid[(size_t)r * N + c];
      if constexpr (OUT_F16)
        ((_Float16*)outp)[(size_t)r * N + c] = (_Float16)v;
      else
        ((float*)outp)[(size_t)r * N + c] = v;
    }
  }
}

// helper: stage a 64x64 f16 tile (row stride in elems) manually (no-TDM fallback)
__device__ __forceinline__ void stage_tile64(const _Float16* g, _Float16* s,
                                             size_t stride) {
  const int tid = threadIdx.x;
#pragma unroll
  for (int i = 0; i < 16; ++i) {
    const int e = tid + i * 256;
    const int r = e >> 6, d = e & 63;
    s[e] = g[(size_t)r * stride + d];
  }
}

// ------------- Local (within-segment) attention per (b, ns, h) -------------
// logits = q*k^T (unmasked, stored f32 for joint softmax later), softmax,
// row_attn = probs * v.
__global__ __launch_bounds__(256) void local_attn_kernel(
    const _Float16* __restrict__ q, const _Float16* __restrict__ k,
    const _Float16* __restrict__ v, float* __restrict__ logits_out,
    _Float16* __restrict__ row_attn) {
  __shared__ _Float16 qs[64 * 64], ks[64 * 64], vs[64 * 64], ps[64 * 64];
  __shared__ float lg[64 * 64];
  const int bid = blockIdx.x;  // = (b*NS + ns)*H + h
  const int h = bid % HQ;
  const int ns = (bid / HQ) % NSQ;
  const int b = bid / (HQ * NSQ);
  const size_t rowbase = (size_t)(b * NSQ + ns) * SEGQ;
  const int tid = threadIdx.x, lane = tid & 31, w = tid >> 5;
  const size_t gbase = rowbase * DQ + h * HDQ;

#if HAVE_TDM
  if (tid < 32) {
    tdm_load_2d_f16(q + gbase, qs, 64, 64, DQ);
    tdm_load_2d_f16(k + gbase, ks, 64, 64, DQ);
    tdm_load_2d_f16(v + gbase, vs, 64, 64, DQ);
    TDM_WAIT0();
  }
#else
  stage_tile64(q + gbase, qs, DQ);
  stage_tile64(k + gbase, ks, DQ);
  stage_tile64(v + gbase, vs, DQ);
#endif
  __syncthreads();

  const int n = lane & 15, mb = (lane >> 4) * 8;
  const size_t lbase = (size_t)bid * SEGQ * SEGQ;
#pragma unroll
  for (int it = 0; it < 2; ++it) {
    const int tt = w * 2 + it, mt = tt >> 2, nt = tt & 3;
    v8f acc = {};
#pragma unroll
    for (int kk = 0; kk < HDQ; kk += 32) {
      const v16h a = load_a_frag(qs + mt * 16 * 64 + kk, 64);
      const v16h bf = load_bT_frag(ks + nt * 16 * 64 + kk, 64);
      acc = wmma16(a, bf, acc);
    }
#pragma unroll
    for (int i = 0; i < 8; ++i) {
      const int r = mt * 16 + mb + i, c = nt * 16 + n;
      lg[r * 64 + c] = acc[i];
      logits_out[lbase + r * 64 + c] = acc[i];
    }
  }
  __syncthreads();

  if (tid < 64) {  // row softmax (unmasked, per reference row_probs)
    const int r = tid;
    float m = -3.4e38f;
    for (int c = 0; c < 64; ++c) m = fmaxf(m, lg[r * 64 + c]);
    float s = 0.f;
    for (int c = 0; c < 64; ++c) {
      const float e = __expf(lg[r * 64 + c] - m);
      lg[r * 64 + c] = e;
      s += e;
    }
    const float inv = 1.f / s;
    for (int c = 0; c < 64; ++c) ps[r * 64 + c] = (_Float16)(lg[r * 64 + c] * inv);
  }
  __syncthreads();

#pragma unroll
  for (int it = 0; it < 2; ++it) {
    const int tt = w * 2 + it, mt = tt >> 2, nt = tt & 3;
    v8f acc = {};
#pragma unroll
    for (int kk = 0; kk < SEGQ; kk += 32) {
      const v16h a = load_a_frag(ps + mt * 16 * 64 + kk, 64);
      const v16h bf = load_b_frag(vs + kk * 64 + nt * 16, 64);
      acc = wmma16(a, bf, acc);
    }
#pragma unroll
    for (int i = 0; i < 8; ++i) {
      const int r = mt * 16 + mb + i, c = nt * 16 + n;
      row_attn[(rowbase + r) * DQ + h * HDQ + c] = (_Float16)acc[i];
    }
  }
}

// ------------------------------ LayerNorm per row ------------------------------
__global__ __launch_bounds__(256) void layernorm_kernel(
    const float* __restrict__ x, const float* __restrict__ sc,
    const float* __restrict__ bi, _Float16* __restrict__ out) {
  __shared__ float red[256];
  const size_t row = blockIdx.x;
  const int tid = threadIdx.x;
  float loc[4];
  float s = 0.f;
#pragma unroll
  for (int i = 0; i < 4; ++i) {
    loc[i] = x[row * DQ + tid + i * 256];
    s += loc[i];
  }
  red[tid] = s;
  __syncthreads();
  for (int off = 128; off > 0; off >>= 1) {
    if (tid < off) red[tid] += red[tid + off];
    __syncthreads();
  }
  const float mu = red[0] * (1.f / DQ);
  __syncthreads();
  float vs = 0.f;
#pragma unroll
  for (int i = 0; i < 4; ++i) {
    const float d = loc[i] - mu;
    vs += d * d;
  }
  red[tid] = vs;
  __syncthreads();
  for (int off = 128; off > 0; off >>= 1) {
    if (tid < off) red[tid] += red[tid + off];
    __syncthreads();
  }
  const float var = red[0] * (1.f / DQ);
  const float inv = rsqrtf(var + 1e-6f);
#pragma unroll
  for (int i = 0; i < 4; ++i) {
    const int c = tid + i * 256;
    out[row * DQ + c] = (_Float16)((loc[i] - mu) * inv * sc[c] + bi[c]);
  }
}

// ------- Joint softmax + global merge per (b, l, h): attend across segments -------
#define NEGCLAMP (-60000.0f)  // representable in f16; exp() underflows to 0
__global__ __launch_bounds__(256) void joint_attn_kernel(
    const _Float16* __restrict__ q, const _Float16* __restrict__ kr,
    const float* __restrict__ llog, const _Float16* __restrict__ ra,
    _Float16* __restrict__ latt, float* __restrict__ mergedg) {
  __shared__ _Float16 qs[64 * 64], ks[64 * 64], ras[64 * 64];
  __shared__ _Float16 gl[64 * 64], ll[64 * 64];
  __shared__ _Float16 lt[64 * 64], gt[64 * 64];
  const int bid = blockIdx.x;  // = (b*SEG + l)*H + h
  const int h = bid % HQ;
  const int l = (bid / HQ) % SEGQ;
  const int b = bid / (HQ * SEGQ);
  const int tid = threadIdx.x, lane = tid & 31, w = tid >> 5;
  const size_t gbase = ((size_t)(b * NSQ) * SEGQ + l) * DQ + h * HDQ;
  const size_t gstride = (size_t)SEGQ * DQ;  // segment stride at fixed l

#if HAVE_TDM
  if (tid < 32) {
    tdm_load_2d_f16(q + gbase, qs, 64, 64, gstride);
    tdm_load_2d_f16(kr + gbase, ks, 64, 64, gstride);
    tdm_load_2d_f16(ra + gbase, ras, 64, 64, gstride);
    TDM_WAIT0();
  }
#else
  stage_tile64(q + gbase, qs, gstride);
  stage_tile64(kr + gbase, ks, gstride);
  stage_tile64(ra + gbase, ras, gstride);
#endif
#pragma unroll
  for (int i = 0; i < 16; ++i) {  // masked local logits (mask: l < c)
    const int e = tid + i * 256;
    const int r = e >> 6, c = e & 63;
    const float vv =
        llog[(((size_t)(b * NSQ + r) * HQ + h) * SEGQ + l) * SEGQ + c];
    ll[e] = (_Float16)((l < c) ? NEGCLAMP : vv);
  }
  __syncthreads();

  const int n = lane & 15, mb = (lane >> 4) * 8;
#pragma unroll
  for (int it = 0; it < 2; ++it) {  // global logits, mask qs<=ks
    const int tt = w * 2 + it, mt = tt >> 2, nt = tt & 3;
    v8f acc = {};
#pragma unroll
    for (int kk = 0; kk < HDQ; kk += 32) {
      const v16h a = load_a_frag(qs + mt * 16 * 64 + kk, 64);
      const v16h bf = load_bT_frag(ks + nt * 16 * 64 + kk, 64);
      acc = wmma16(a, bf, acc);
    }
#pragma unroll
    for (int i = 0; i < 8; ++i) {
      const int r = mt * 16 + mb + i, c = nt * 16 + n;
      gl[r * 64 + c] = (_Float16)((r <= c) ? NEGCLAMP : acc[i]);
    }
  }
  __syncthreads();

  if (tid < 64) {  // joint softmax over [local(64) | global(64)]
    const int r = tid;
    float m = -3.4e38f;
    for (int c = 0; c < 64; ++c) m = fmaxf(m, (float)ll[r * 64 + c]);
    for (int c = 0; c < 64; ++c) m = fmaxf(m, (float)gl[r * 64 + c]);
    float s = 0.f;
    float el[64], eg[64];
    for (int c = 0; c < 64; ++c) {
      el[c] = __expf((float)ll[r * 64 + c] - m);
      s += el[c];
    }
    for (int c = 0; c < 64; ++c) {
      eg[c] = __expf((float)gl[r * 64 + c] - m);
      s += eg[c];
    }
    const float inv = 1.f / s;
    for (int c = 0; c < 64; ++c) {
      const _Float16 pl = (_Float16)(el[c] * inv);
      lt[r * 64 + c] = pl;
      latt[(((size_t)(b * NSQ + r) * HQ + h) * SEGQ + l) * SEGQ + c] = pl;
      gt[r * 64 + c] = (_Float16)(eg[c] * inv);
    }
  }
  __syncthreads();
  (void)lt;

#pragma unroll
  for (int it = 0; it < 2; ++it) {  // global merge: gatt(qs x vs) * row_attn(vs x d)
    const int tt = w * 2 + it, mt = tt >> 2, nt = tt & 3;
    v8f acc = {};
#pragma unroll
    for (int kk = 0; kk < NSQ; kk += 32) {
      const v16h a = load_a_frag(gt + mt * 16 * 64 + kk, 64);
      const v16h bf = load_b_frag(ras + kk * 64 + nt * 16, 64);
      acc = wmma16(a, bf, acc);
    }
#pragma unroll
    for (int i = 0; i < 8; ++i) {
      const int r = mt * 16 + mb + i, c = nt * 16 + n;
      mergedg[((size_t)(b * NSQ + r) * SEGQ + l) * DQ + h * HDQ + c] = acc[i];
    }
  }
}

// ------ Local merge per (b, ns, h): latt * v, add global part, -> f16 merged ------
__global__ __launch_bounds__(256) void local_merge_kernel(
    const _Float16* __restrict__ latt, const _Float16* __restrict__ v,
    const float* __restrict__ mergedg, _Float16* __restrict__ merged) {
  __shared__ _Float16 lts[64 * 64], vs[64 * 64];
  const int bid = blockIdx.x;  // = (b*NS + ns)*H + h
  const int h = bid % HQ;
  const int ns = (bid / HQ) % NSQ;
  const int b = bid / (HQ * NSQ);
  const size_t rowbase = (size_t)(b * NSQ + ns) * SEGQ;
  const int tid = threadIdx.x, lane = tid & 31, w = tid >> 5;

#if HAVE_TDM
  if (tid < 32) {
    tdm_load_2d_f16(latt + (size_t)bid * SEGQ * SEGQ, lts, 4096, 1, 4096);
    tdm_load_2d_f16(v + rowbase * DQ + h * HDQ, vs, 64, 64, DQ);
    TDM_WAIT0();
  }
#else
  {
    const int t2 = threadIdx.x;
#pragma unroll
    for (int i = 0; i < 16; ++i) {
      const int e = t2 + i * 256;
      lts[e] = latt[(size_t)bid * SEGQ * SEGQ + e];
    }
  }
  stage_tile64(v + rowbase * DQ + h * HDQ, vs, DQ);
#endif
  __syncthreads();

  const int n = lane & 15, mb = (lane >> 4) * 8;
#pragma unroll
  for (int it = 0; it < 2; ++it) {
    const int tt = w * 2 + it, mt = tt >> 2, nt = tt & 3;
    v8f acc = {};
#pragma unroll
    for (int kk = 0; kk < SEGQ; kk += 32) {
      const v16h a = load_a_frag(lts + mt * 16 * 64 + kk, 64);
      const v16h bf = load_b_frag(vs + kk * 64 + nt * 16, 64);
      acc = wmma16(a, bf, acc);
    }
#pragma unroll
    for (int i = 0; i < 8; ++i) {
      const int r = mt * 16 + mb + i, c = nt * 16 + n;
      const size_t idx = (rowbase + r) * DQ + h * HDQ + c;
      merged[idx] = (_Float16)(acc[i] + mergedg[idx]);
    }
  }
}

// ----------------------------------- launch -----------------------------------
extern "C" void kernel_launch(void* const* d_in, const int* in_sizes, int n_in,
                              void* d_out, int out_size, void* d_ws,
                              size_t ws_size, hipStream_t stream) {
  const float* x    = (const float*)d_in[0];
  const float* Wq   = (const float*)d_in[1];
  const float* Wk   = (const float*)d_in[2];
  const float* Wv   = (const float*)d_in[3];
  const float* Wro  = (const float*)d_in[4];
  const float* lns  = (const float*)d_in[5];
  const float* lnb  = (const float*)d_in[6];
  const float* Wk2  = (const float*)d_in[7];
  const float* Wout = (const float*)d_in[8];
  float* out = (float*)d_out;

  // Workspace layout (bytes); total = NELQ*22 + 12MB ~= 364 MB.
  char* ws = (char*)d_ws;
  _Float16* qf    = (_Float16*)(ws);                 // f16  [0,   2N)
  _Float16* kf    = (_Float16*)(ws + NELQ * 2);      // f16  [2N,  4N)
  _Float16* vf    = (_Float16*)(ws + NELQ * 4);      // f16  [4N,  6N)
  float*    llog  = (float*)   (ws + NELQ * 6);      // f32  [6N, 10N)
  _Float16* ra    = (_Float16*)(ws + NELQ * 10);     // f16  [10N,12N)
  float*    kraw  = (float*)   (ws + NELQ * 12);     // f32  [12N,16N)
  _Float16* kln   = (_Float16*)(ws + NELQ * 16);     // f16  [16N,18N)
  _Float16* kproj = (_Float16*)(ws + NELQ * 18);     // f16  [18N,20N)
  _Float16* xh    = (_Float16*)(ws + NELQ * 20);     // f16  [20N,22N)
  const size_t WB = (size_t)DQ * DQ;                 // 1M elems per weight
  _Float16* wh[6];
  for (int j = 0; j < 6; ++j) wh[j] = (_Float16*)(ws + NELQ * 22 + j * WB * 2);
  _Float16* latt  = kf;    // reuse: k dead after local_attn
  float*    mg    = kraw;  // reuse: key_row_raw dead after layernorm
  _Float16* mrg   = kln;   // reuse: key_row_ln dead after k2 projection

  const dim3 gemm_grid(DQ / 64, BSQ / 128, 1);
  const dim3 blk(256, 1, 1);

  // 0) One-time f32 -> f16 conversions (x + 6 weight matrices)
  cvt_f16_kernel<<<dim3((unsigned)(NELQ / 256)), blk, 0, stream>>>(x, xh, (int)NELQ);
  const float* wsrc[6] = {Wq, Wk, Wv, Wro, Wk2, Wout};
  for (int j = 0; j < 6; ++j)
    cvt_f16_kernel<<<dim3((unsigned)(WB / 256)), blk, 0, stream>>>(wsrc[j], wh[j], (int)WB);

  // 1) QKV projections (q pre-scaled by 1/sqrt(HD) = 0.125)
  gemm_kernel<true, false><<<gemm_grid, blk, 0, stream>>>(
      xh, wh[0], qf, nullptr, 0.125f, DQ, DQ);
  gemm_kernel<true, false><<<gemm_grid, blk, 0, stream>>>(
      xh, wh[1], kf, nullptr, 1.0f, DQ, DQ);
  gemm_kernel<true, false><<<gemm_grid, blk, 0, stream>>>(
      xh, wh[2], vf, nullptr, 1.0f, DQ, DQ);

  // 2) Local attention per (b, ns, h)
  local_attn_kernel<<<dim3(BQ * NSQ * HQ), blk, 0, stream>>>(qf, kf, vf, llog, ra);

  // 3) row_attn @ Wrow_out + x -> f32, then LayerNorm -> f16
  gemm_kernel<false, true><<<gemm_grid, blk, 0, stream>>>(
      ra, wh[3], kraw, x, 1.0f, DQ, DQ);
  layernorm_kernel<<<dim3(BSQ), blk, 0, stream>>>(kraw, lns, lnb, kln);

  // 4) key_row @ Wk2 -> f16
  gemm_kernel<true, false><<<gemm_grid, blk, 0, stream>>>(
      kln, wh[4], kproj, nullptr, 1.0f, DQ, DQ);

  // 5) Joint softmax + global merge per (b, l, h)
  joint_attn_kernel<<<dim3(BQ * SEGQ * HQ), blk, 0, stream>>>(
      qf, kproj, llog, ra, latt, mg);

  // 6) Local merge + combine per (b, ns, h)
  local_merge_kernel<<<dim3(BQ * NSQ * HQ), blk, 0, stream>>>(latt, vf, mg, mrg);

  // 7) Output projection merged @ Wout -> f32 out
  gemm_kernel<false, false><<<gemm_grid, blk, 0, stream>>>(
      mrg, wh[5], out, nullptr, 1.0f, DQ, DQ);
}
